// PathMultiheadAttention_41420664603051
// MI455X (gfx1250) — compile-verified, matrix operand order checked by reading, not run on hardware
//
#include <hip/hip_runtime.h>
#include <hip/hip_bf16.h>

// ---------------- types ----------------
typedef __attribute__((ext_vector_type(16))) _Float16 h16;
typedef __attribute__((ext_vector_type(8)))  _Float16 h8;
typedef __attribute__((ext_vector_type(4)))  _Float16 h4;
typedef __attribute__((ext_vector_type(8)))  float    f8;
typedef __attribute__((ext_vector_type(4)))  unsigned int u32x4;
typedef __attribute__((ext_vector_type(4)))  int i32x4;
typedef __attribute__((ext_vector_type(8)))  int i32x8;

#define EMBED 1024
#define HEADS 16
#define HD    64
#define SEQ   2048
#define BATCH 2

#if defined(__has_builtin)
#if __has_builtin(__builtin_amdgcn_tensor_load_to_lds)
#define HAVE_TDM 1
#endif
#endif
#ifndef HAVE_TDM
#define HAVE_TDM 0
#endif

// D = A(16x32) * B(32x16)^T-form + C, f16 in, f32 acc
__device__ __forceinline__ f8 wmma_f16(h16 a, h16 b, f8 c) {
  return __builtin_amdgcn_wmma_f32_16x16x32_f16(false, a, false, b, (short)0, c,
                                                false, false);
}

// Fragment gather for V_WMMA_F32_16X16X32_F16 from a row-major f16 tile:
// lane L holds row (caller-supplied); halves 0..7 = k base..base+7,
// halves 8..15 = k base+16..base+23, base = kbase + (L>=16 ? 8 : 0).
__device__ __forceinline__ h16 frag_ld(const _Float16* p, int rstride, int row,
                                       int kbase, int lane) {
  const int klo = kbase + ((lane & 16) ? 8 : 0);
  const _Float16* q = p + row * rstride + klo;
  h8 lo = *(const h8*)(q);
  h8 hi = *(const h8*)(q + 16);
  h16 f;
#pragma unroll
  for (int i = 0; i < 8; ++i) { f[i] = lo[i]; f[i + 8] = hi[i]; }
  return f;
}

template <int N>
__device__ __forceinline__ void wait_tensorcnt() {
#if defined(__has_builtin) && __has_builtin(__builtin_amdgcn_s_wait_tensorcnt)
  __builtin_amdgcn_s_wait_tensorcnt(N);
#else
  asm volatile("s_wait_tensorcnt %0" ::"i"(N) : "memory");
#endif
}

#if HAVE_TDM
// TDM 2D tile load: rows x row_elems f16 elements, global row stride
// row_stride elems, into LDS at lds_ptr with per-row padding.
// pad_int: dwords-before-pad code (0=2,1=4,2=8,3=16,4=32);
// pad_amt: pad dwords minus 1. Descriptor per CDNA5 ISA s8.
// The LDS byte address is the low 32 bits of the flat pointer (ISA 10.2
// aperture rule); routing the real pointer in (plus the asm escape below)
// makes the DMA's LDS write visible to alias analysis so the consuming
// ds_load fragments are not folded to undef.
__device__ __forceinline__ void tdm_load_2d(_Float16* lds_ptr,
                                            unsigned long long gaddr,
                                            unsigned int row_elems,
                                            unsigned int rows,
                                            unsigned int row_stride,
                                            unsigned int pad_int,
                                            unsigned int pad_amt) {
  const unsigned int lds_off = (unsigned int)(uintptr_t)lds_ptr;
  u32x4 g0;
  g0[0] = 1u;                                       // count=1, user mode
  g0[1] = lds_off;                                  // lds_addr (bytes)
  g0[2] = (unsigned int)gaddr;                      // global_addr[31:0]
  g0[3] = (unsigned int)(gaddr >> 32) | (2u << 30); // global_addr hi, type=2
  i32x8 g1;
  g1[0] = (int)((1u << 16) |                        // data_size = 2 bytes
                (1u << 20) |                        // pad_enable
                (pad_int << 22) | (pad_amt << 25));
  g1[1] = (int)((row_elems & 0xFFFFu) << 16);       // tensor_dim0[15:0]
  g1[2] = (int)((row_elems >> 16) | ((rows & 0xFFFFu) << 16));
  g1[3] = (int)((rows >> 16) | ((row_elems & 0xFFFFu) << 16));  // | tile_dim0
  g1[4] = (int)(rows & 0xFFFFu);                    // tile_dim1 (tile_dim2=0)
  g1[5] = (int)row_stride;                          // tensor_dim0_stride lo32
  g1[6] = 0;
  g1[7] = 0;
  i32x4 z4 = {0, 0, 0, 0};
#if defined(__clang_major__) && (__clang_major__ >= 23)
  i32x8 z8 = {0, 0, 0, 0, 0, 0, 0, 0};
  __builtin_amdgcn_tensor_load_to_lds(g0, g1, z4, z4, z8, 0);
#else
  __builtin_amdgcn_tensor_load_to_lds(g0, g1, z4, z4, 0);
#endif
  // Escape hatch: the DMA writes through lds_ptr; model it as a clobber.
  asm volatile("" ::"r"(lds_ptr) : "memory");
}
#endif

// ---------------------------------------------------------------------------
// Kernel 0: one-time fp32 -> f16 conversion (memory bound, ~2us at 23.3TB/s).
// ---------------------------------------------------------------------------
__global__ __launch_bounds__(256)
void cvt_f16(const float* __restrict__ src, _Float16* __restrict__ dst, int n) {
  const int i = (blockIdx.x * 256 + threadIdx.x) * 8;
  if (i < n) {
    float4 a = *(const float4*)(src + i);
    float4 b = *(const float4*)(src + i + 4);
    h8 o = {(_Float16)a.x, (_Float16)a.y, (_Float16)a.z, (_Float16)a.w,
            (_Float16)b.x, (_Float16)b.y, (_Float16)b.z, (_Float16)b.w};
    *(h8*)(dst + i) = o;
  }
}

// ---------------------------------------------------------------------------
// Shared GEMM main loop: C(128x128) += Ah(128xK) * Bh(128xK)^T, all f16,
// K = EMBED, both tiles DMA'd by the TDM into double-buffered LDS
// (40-half padded rows: 16 data dwords + 4 pad dwords -> pad 3/3).
// Pipelining: issue tile pair i+1, s_wait_tensorcnt<=2 for pair i, barrier,
// WMMA on buffer i while the TDM fills buffer i+1.
// ---------------------------------------------------------------------------
#define GEMM_BUF_H 10240  // halves per LDS buffer (As 5120 + Bs 5120)

__device__ __forceinline__ void gemm_mainloop(const _Float16* Ag,
                                              const _Float16* Bg,
                                              _Float16* smem, int wm, int wn,
                                              int lane, int wid,
                                              f8 acc[2][4]) {
#if HAVE_TDM
  auto issue = [&](int buf, int kb) {
    _Float16* dst = smem + buf * GEMM_BUF_H;
    tdm_load_2d(dst, (unsigned long long)(uintptr_t)(Ag + kb), 32u, 128u,
                (unsigned int)EMBED, 3u, 3u);
    tdm_load_2d(dst + 5120, (unsigned long long)(uintptr_t)(Bg + kb), 32u,
                128u, (unsigned int)EMBED, 3u, 3u);
  };
  if (wid == 0) issue(0, 0);
#endif
  for (int kb = 0, it = 0; kb < EMBED; kb += 32, ++it) {
    const int buf = it & 1;
#if HAVE_TDM
    if (wid == 0) {
      if (kb + 32 < EMBED) {
        issue(buf ^ 1, kb + 32);   // prefetch next pair
        wait_tensorcnt<2>();       // oldest pair (this buffer) complete
      } else {
        wait_tensorcnt<0>();
      }
    }
#else
    {  // fallback: per-thread f16 staging
      const int trow = (threadIdx.x & 255) >> 2;
      const int cg = (threadIdx.x & 3) * 8;
      _Float16* As = smem + buf * GEMM_BUF_H;
      _Float16* Bs = As + 5120;
      *(h8*)(&As[(trow & 127) * 40 + cg]) =
          *(const h8*)(Ag + (size_t)trow * EMBED + kb + cg);
      *(h8*)(&As[((trow + 64) & 127) * 40 + cg]) =
          *(const h8*)(Ag + (size_t)((trow + 64) & 127) * EMBED + kb + cg);
      *(h8*)(&Bs[trow * 40 + cg]) =
          *(const h8*)(Bg + (size_t)trow * EMBED + kb + cg);
      *(h8*)(&Bs[(trow + 64) * 40 + cg]) =
          *(const h8*)(Bg + (size_t)(trow + 64) * EMBED + kb + cg);
    }
#endif
    __syncthreads();
    const _Float16* As = smem + buf * GEMM_BUF_H;
    const _Float16* Bs = As + 5120;
    h16 af[2], bf[4];
#pragma unroll
    for (int mt = 0; mt < 2; ++mt)
      af[mt] = frag_ld(As, 40, wm * 32 + mt * 16 + (lane & 15), 0, lane);
#pragma unroll
    for (int nt = 0; nt < 4; ++nt)
      bf[nt] = frag_ld(Bs, 40, wn * 64 + nt * 16 + (lane & 15), 0, lane);
#pragma unroll
    for (int mt = 0; mt < 2; ++mt)
#pragma unroll
      for (int nt = 0; nt < 4; ++nt)
        acc[mt][nt] = wmma_f16(af[mt], bf[nt], acc[mt][nt]);
    __syncthreads();
  }
}

// ---------------------------------------------------------------------------
// Kernel 1: QKV = xh @ w_attn_h^T + b_attn   (f16 WMMA, fp32 acc)
// Writes Q[b][h][s][hd], K[b][h][s][hd], Vt[b][h][hd][s] as f16.
// Grid: (3072/128, 4096/128), block 256 (8 waves, 4x2, each wave 32x64).
// ---------------------------------------------------------------------------
__global__ __launch_bounds__(256)
void qkv_gemm(const _Float16* __restrict__ Xh, const _Float16* __restrict__ Wh,
              const float* __restrict__ bias, _Float16* __restrict__ Q,
              _Float16* __restrict__ K, _Float16* __restrict__ Vt) {
  __shared__ __align__(16) _Float16 smem[2 * GEMM_BUF_H];  // 40KB, 2 buffers
  const int t = threadIdx.x;
  const int lane = t & 31, wid = t >> 5;
  const int wm = wid & 3, wn = wid >> 2;
  const int m0 = blockIdx.y * 128, n0 = blockIdx.x * 128;

  f8 acc[2][4] = {};
  gemm_mainloop(Xh + (size_t)m0 * EMBED, Wh + (size_t)n0 * EMBED, smem, wm, wn,
                lane, wid, acc);

  // Epilogue: bias + f16 scatter. A 128-wide N tile never straddles the
  // Q/K/V boundaries (multiples of 1024) -> single uniform branch.
  const int roff = (lane & 16) ? 8 : 0;
  const int nlo = lane & 15;
  const int seg = n0 >> 10;  // 0=Q 1=K 2=V (wave-uniform)
  const int c0 = n0 & (EMBED - 1);
  _Float16* dstQK = (seg == 0) ? Q : K;
#pragma unroll
  for (int mt = 0; mt < 2; ++mt) {
#pragma unroll
    for (int nt = 0; nt < 4; ++nt) {
#pragma unroll
      for (int j = 0; j < 8; ++j) {
        const int r = m0 + wm * 32 + mt * 16 + j + roff;
        const int c = c0 + wn * 64 + nt * 16 + nlo;  // column within segment
        const float v = acc[mt][nt][j] + bias[seg * EMBED + c];
        const int b = r >> 11, s = r & (SEQ - 1);
        const int h = c >> 6, d = c & 63;
        const _Float16 hv = (_Float16)v;
        if (seg < 2) {
          dstQK[((size_t)((b * HEADS + h) * SEQ + s) << 6) + d] = hv;
        } else {
          Vt[(size_t)((b * HEADS + h) * HD + d) * SEQ + s] = hv;
        }
      }
    }
  }
}

// ---------------------------------------------------------------------------
// Kernel 2: flash attention (no scale, no mask, softmax over keys).
// Grid: (B*H, SEQ/64), block 128 (4 waves, each owns 16 q-rows x full hd).
// K/V^T tiles streamed by the TDM into double-buffered LDS (72-half padded
// rows: 32 data dwords + 4 pad dwords -> pad 4/3), TENSORcnt-pipelined.
// ---------------------------------------------------------------------------
#define KV_BUF_H 9216  // halves per LDS buffer (Ks 4608 + Vs 4608)

__global__ __launch_bounds__(128)
void flash_attn(const _Float16* __restrict__ Q, const _Float16* __restrict__ K,
                const _Float16* __restrict__ Vt, _Float16* __restrict__ A) {
  __shared__ __align__(16) _Float16 smem[2 * KV_BUF_H + 4 * 16 * 72];
  _Float16* Ps = smem + 2 * KV_BUF_H;  // per-wave P tiles
  const int t = threadIdx.x;
  const int lane = t & 31, wid = t >> 5;
  const int bh = blockIdx.x;                  // b*HEADS + h
  const int q0 = blockIdx.y * 64 + wid * 16;  // this wave's first q row

  // Q fragments (16 rows x 64 hd = 2 k-steps) kept in registers.
  h16 aq[2];
#pragma unroll
  for (int ks = 0; ks < 2; ++ks)
    aq[ks] = frag_ld(Q + ((size_t)bh * SEQ + q0) * HD, HD, lane & 15, ks * 32,
                     lane);

  f8 acco[4] = {};
  float mi[8], li[8];
#pragma unroll
  for (int j = 0; j < 8; ++j) { mi[j] = -3.0e38f; li[j] = 0.f; }

  _Float16* myP = Ps + wid * 16 * 72;

#if HAVE_TDM
  auto issue_kv = [&](int buf, int kt) {
    _Float16* dst = smem + buf * KV_BUF_H;
    tdm_load_2d(dst,
                (unsigned long long)(uintptr_t)(K + ((size_t)bh * SEQ + kt) * HD),
                64u, 64u, 64u, 4u, 3u);
    tdm_load_2d(dst + 4608,
                (unsigned long long)(uintptr_t)(Vt + (size_t)bh * HD * SEQ + kt),
                64u, 64u, (unsigned int)SEQ, 4u, 3u);
  };
  if (wid == 0) issue_kv(0, 0);
#endif

  for (int kt = 0, it = 0; kt < SEQ; kt += 64, ++it) {
    const int buf = it & 1;
#if HAVE_TDM
    if (wid == 0) {
      if (kt + 64 < SEQ) {
        issue_kv(buf ^ 1, kt + 64);
        wait_tensorcnt<2>();
      } else {
        wait_tensorcnt<0>();
      }
    }
#else
    {
      _Float16* Ks = smem + buf * KV_BUF_H;
      _Float16* Vs = Ks + 4608;
#pragma unroll
      for (int pass = 0; pass < 4; ++pass) {
        const int r = pass * 16 + (t >> 3);
        const int cg = (t & 7) * 8;
        *(h8*)(&Ks[r * 72 + cg]) =
            *(const h8*)(K + ((size_t)bh * SEQ + kt + r) * HD + cg);
        *(h8*)(&Vs[r * 72 + cg]) =
            *(const h8*)(Vt + ((size_t)bh * HD + r) * SEQ + kt + cg);
      }
    }
#endif
    __syncthreads();
    const _Float16* Ks = smem + buf * KV_BUF_H;
    const _Float16* Vs = Ks + 4608;

    // Scores: S(16x64) = Q(16x64) * K_tile^T
    f8 sc[4] = {};
#pragma unroll
    for (int ks = 0; ks < 2; ++ks)
#pragma unroll
      for (int nt = 0; nt < 4; ++nt)
        sc[nt] = wmma_f16(
            aq[ks], frag_ld(Ks, 72, nt * 16 + (lane & 15), ks * 32, lane),
            sc[nt]);

    // Online softmax: rows j (lanes 0-15) / j+8 (lanes 16-31), cols lane&15.
#pragma unroll
    for (int j = 0; j < 8; ++j) {
      float pm = fmaxf(fmaxf(sc[0][j], sc[1][j]), fmaxf(sc[2][j], sc[3][j]));
#pragma unroll
      for (int off = 1; off < 16; off <<= 1)
        pm = fmaxf(pm, __shfl_xor(pm, off, 16));
      const float nm = fmaxf(mi[j], pm);
      const float scale = __expf(mi[j] - nm);
      float rs = 0.f;
      const int prow = (j + ((lane & 16) ? 8 : 0)) * 72 + (lane & 15);
#pragma unroll
      for (int nt = 0; nt < 4; ++nt) {
        const float p = __expf(sc[nt][j] - nm);
        rs += p;
        myP[prow + nt * 16] = (_Float16)p;  // re-layout P as A-fragment source
      }
#pragma unroll
      for (int off = 1; off < 16; off <<= 1) rs += __shfl_xor(rs, off, 16);
      li[j] = li[j] * scale + rs;
      mi[j] = nm;
#pragma unroll
      for (int nt = 0; nt < 4; ++nt) acco[nt][j] *= scale;
    }

    // O += P(16x64) * V(64x64); V^T layout keeps B-fragments K-contiguous.
#pragma unroll
    for (int ks = 0; ks < 2; ++ks) {
      h16 ap = frag_ld(myP, 72, lane & 15, ks * 32, lane);
#pragma unroll
      for (int nt = 0; nt < 4; ++nt)
        acco[nt] = wmma_f16(
            ap, frag_ld(Vs, 72, nt * 16 + (lane & 15), ks * 32, lane),
            acco[nt]);
    }
    __syncthreads();
  }

  // Epilogue: normalize, store merged-head f16 activations A[b][s][h*64+d].
  const int b = bh >> 4, h = bh & 15;
  const int roff = (lane & 16) ? 8 : 0;
#pragma unroll
  for (int j = 0; j < 8; ++j) {
    const float inv = 1.0f / li[j];
    const int s = q0 + j + roff;
#pragma unroll
    for (int nt = 0; nt < 4; ++nt) {
      const float v = acco[nt][j] * inv;
      A[(size_t)(b * SEQ + s) * EMBED + h * HD + nt * 16 + (lane & 15)] =
          (_Float16)v;
    }
  }
}

// ---------------------------------------------------------------------------
// Kernel 3: out = A @ w_proj_h^T + b_proj  (f16 WMMA via TDM, fp32 out)
// Grid: (1024/128, 4096/128), block 256.
// ---------------------------------------------------------------------------
__global__ __launch_bounds__(256)
void proj_gemm(const _Float16* __restrict__ Af, const _Float16* __restrict__ Wh,
               const float* __restrict__ bias, float* __restrict__ Out) {
  __shared__ __align__(16) _Float16 smem[2 * GEMM_BUF_H];
  const int t = threadIdx.x;
  const int lane = t & 31, wid = t >> 5;
  const int wm = wid & 3, wn = wid >> 2;
  const int m0 = blockIdx.y * 128, n0 = blockIdx.x * 128;

  f8 acc[2][4] = {};
  gemm_mainloop(Af + (size_t)m0 * EMBED, Wh + (size_t)n0 * EMBED, smem, wm, wn,
                lane, wid, acc);

  const int roff = (lane & 16) ? 8 : 0;
  const int nlo = lane & 15;
#pragma unroll
  for (int mt = 0; mt < 2; ++mt)
#pragma unroll
    for (int nt = 0; nt < 4; ++nt)
#pragma unroll
      for (int j = 0; j < 8; ++j) {
        const int r = m0 + wm * 32 + mt * 16 + j + roff;
        const int c = n0 + wn * 64 + nt * 16 + nlo;
        Out[(size_t)r * EMBED + c] = acc[mt][nt][j] + bias[c];
      }
}

// ---------------------------------------------------------------------------
extern "C" void kernel_launch(void* const* d_in, const int* in_sizes, int n_in,
                              void* d_out, int out_size, void* d_ws,
                              size_t ws_size, hipStream_t stream) {
  (void)in_sizes; (void)n_in; (void)out_size; (void)ws_size;
  const float* x      = (const float*)d_in[0];
  const float* w_attn = (const float*)d_in[1];
  const float* b_attn = (const float*)d_in[2];
  const float* w_proj = (const float*)d_in[3];
  const float* b_proj = (const float*)d_in[4];
  float* out = (float*)d_out;

  const size_t NX = (size_t)BATCH * SEQ * EMBED;         // 4M
  const size_t NWA = (size_t)3 * EMBED * EMBED;          // 3M
  const size_t NWP = (size_t)EMBED * EMBED;              // 1M
  const size_t HBUF = (size_t)BATCH * HEADS * SEQ * HD;  // 4M

  _Float16* Xh  = (_Float16*)d_ws;    // f16 copies of inputs
  _Float16* Wah = Xh + NX;
  _Float16* Wph = Wah + NWA;
  _Float16* Q   = Wph + NWP;
  _Float16* K   = Q + HBUF;
  _Float16* Vt  = K + HBUF;
  _Float16* A   = Vt + HBUF;          // merged-head activations [B*S][EMBED]

  cvt_f16<<<NX / (8 * 256), 256, 0, stream>>>(x, Xh, (int)NX);
  cvt_f16<<<NWA / (8 * 256), 256, 0, stream>>>(w_attn, Wah, (int)NWA);
  cvt_f16<<<NWP / (8 * 256), 256, 0, stream>>>(w_proj, Wph, (int)NWP);

  qkv_gemm<<<dim3(3 * EMBED / 128, BATCH * SEQ / 128), 256, 0, stream>>>(
      Xh, Wah, b_attn, Q, K, Vt);
  flash_attn<<<dim3(BATCH * HEADS, SEQ / 64), 128, 0, stream>>>(Q, K, Vt, A);
  proj_gemm<<<dim3(EMBED / 128, BATCH * SEQ / 128), 256, 0, stream>>>(
      A, Wph, b_proj, out);
}
